// TransformerDecoder_50173807952576
// MI455X (gfx1250) — compile-verified, matrix operand order checked by reading, not run on hardware
//
#include <hip/hip_runtime.h>
#include <hip/hip_bf16.h>

typedef __bf16 bf16;
typedef __attribute__((ext_vector_type(16))) __bf16 v16bf;
typedef __attribute__((ext_vector_type(8)))  float  v8f;

#define BB   16
#define SQL  256
#define SEL  196
#define SEP  224
#define HD   512
#define NHD  8
#define NLAY 4
#define VOC  16000
#define TT   (BB * SQL)      // 4096 tokens
#define ZB   (BB * NHD)      // 128 batched (b,n) slices
#define SLOT (256 * HD)      // per-(b,n) q/k/v slot (elems)

#define BM 128
#define BN 128
#define BK 32
#define AST 40    // LDS stride (halves) for A tile rows (32 + 8 pad) -> 80B, 16B aligned
#define BSN 136   // LDS stride for normal-B [k][n] layout (128 + 8 pad)
#define BST 40    // LDS stride for trans-B  [n][k] layout (32 + 8 pad)

// Async global->LDS staging (GLOBAL_LOAD_ASYNC_TO_LDS_B128, ASYNCcnt-tracked).
// The clang builtin's pointer params live in the unspellable __device__ LangAS,
// so issue the instruction via inline asm. LDS operand = low 32 bits of the
// generic shared pointer (flat LDS addresses are wave-relative addr[31:0]).
#if defined(__gfx1250__) && __has_builtin(__builtin_amdgcn_s_wait_asynccnt)
#define ASYNC_COPY 1
#define ASYNC_B128(lp, gp)                                                   \
    asm volatile("global_load_async_to_lds_b128 %0, %1, off"                 \
                 :: "v"((unsigned)(unsigned long long)(lp)),                 \
                    "v"((unsigned long long)(gp))                            \
                 : "memory")
#else
#define ASYNC_COPY 0
#endif

struct GemmP {
    const bf16* A; const bf16* B; const float* bias;
    float* Cf; bf16* Cb;
    int M, N, K, lda, ldb, ldc, zdiv, transB, relu;
    long long aZd, aZm, bZd, bZm, cZd, cZm, biasZd, biasZm;
    float alpha;
};

union FragU { v16bf v; uint4 q[2]; };

// ---------------------------------------------------------------- GEMM (bf16 WMMA)
// 128x128x32 block tiles, 8 waves (2x4), 8 v_wmma per K-step per wave,
// double-buffered LDS with async copy overlapped against the WMMA burst.
__global__ __launch_bounds__(256) void gemm_bf16_kernel(GemmP p) {
    __shared__ bf16 As[2][BM * AST];                    // 2 x 10240 B
    __shared__ bf16 Bs[2][BM * BST];                    // 2 x 10240 B (covers both layouts)

    const int tid   = threadIdx.x;
    const int lane  = tid & 31;
    const int wid   = tid >> 5;
    const int waveM = wid & 1;              // 2 waves along M (64 rows each)
    const int waveN = wid >> 1;             // 4 waves along N (32 cols each)
    const int lm    = lane & 15;
    const int hi    = lane >> 4;
    const int kb    = hi * 8;

    const int z = blockIdx.z;
    const long long zq = z / p.zdiv, zr = z % p.zdiv;
    const bf16* Az = p.A + zq * p.aZd + zr * p.aZm;
    const bf16* Bz = p.B + zq * p.bZd + zr * p.bZm;
    const long long coff = zq * p.cZd + zr * p.cZm;
    const float* biasz = p.bias ? (p.bias + zq * p.biasZd + zr * p.biasZm) : (const float*)0;

    const int m0 = blockIdx.x * BM;
    const int n0 = blockIdx.y * BN;

    // stage one 128x32 A tile + one B tile into LDS buffer `buf`
    auto stage = [&](int buf, int k0) {
        bf16* AsB = &As[buf][0];
        bf16* BsB = &Bs[buf][0];
#pragma unroll
        for (int c = 0; c < 2; ++c) {
            int ch = tid + c * 256;
            int r  = ch >> 2, cc = (ch & 3) * 8;
            const bf16* gp = Az + (long long)(m0 + r) * p.lda + k0 + cc;
            bf16* lp = AsB + r * AST + cc;
#if ASYNC_COPY
            ASYNC_B128(lp, gp);
#else
            *(uint4*)lp = *(const uint4*)gp;
#endif
        }
#pragma unroll
        for (int c = 0; c < 2; ++c) {
            int ch = tid + c * 256;
            const bf16* gp; bf16* lp;
            if (!p.transB) {                 // B is (K,N): tile 32x128 as [k][n]
                int kr = ch >> 4, nc = (ch & 15) * 8;
                gp = Bz + (long long)(k0 + kr) * p.ldb + n0 + nc;
                lp = BsB + kr * BSN + nc;
            } else {                          // B is (N,K): tile 128x32 as [n][k]
                int nr = ch >> 2, kc = (ch & 3) * 8;
                gp = Bz + (long long)(n0 + nr) * p.ldb + k0 + kc;
                lp = BsB + nr * BST + kc;
            }
#if ASYNC_COPY
            ASYNC_B128(lp, gp);
#else
            *(uint4*)lp = *(const uint4*)gp;
#endif
        }
    };

    v8f acc[4][2];
#pragma unroll
    for (int a = 0; a < 4; ++a)
#pragma unroll
        for (int b = 0; b < 2; ++b)
#pragma unroll
            for (int r = 0; r < 8; ++r) acc[a][b][r] = 0.0f;

    const int nk = p.K / BK;
    stage(0, 0);                              // prologue: tile 0 in flight

    for (int ki = 0; ki < nk; ++ki) {
        const int cur = ki & 1;
#if ASYNC_COPY
        __builtin_amdgcn_s_wait_asynccnt(0);  // own copies for tile ki complete
#endif
        __syncthreads();                      // all waves' copies visible
        if (ki + 1 < nk) stage(cur ^ 1, (ki + 1) * BK);   // overlap next copy with compute
        if ((long long)(ki + 2) * BK < p.K)   // warm L2 two tiles ahead
            __builtin_prefetch(Az + (long long)(m0 + (tid >> 1)) * p.lda + (ki + 2) * BK, 0, 3);

        // ---- build fragments (16-bit A 16x32 lane striping per ISA 7.12.2)
        const bf16* AsC = &As[cur][0];
        const bf16* BsC = &Bs[cur][0];
        v16bf afr[4], bfr[2];
#pragma unroll
        for (int sm = 0; sm < 4; ++sm) {
            const bf16* ap = AsC + (waveM * 64 + sm * 16 + lm) * AST;
            FragU u;
            u.q[0] = *(const uint4*)(ap + kb);        // K = kb .. kb+7
            u.q[1] = *(const uint4*)(ap + kb + 16);   // K = kb+16 .. kb+23
            afr[sm] = u.v;
        }
        if (!p.transB) {
#pragma unroll
            for (int sn = 0; sn < 2; ++sn) {
                int n = waveN * 32 + sn * 16 + lm;
#pragma unroll
                for (int e = 0; e < 8; ++e) {
                    bfr[sn][e]     = BsC[(kb + e) * BSN + n];
                    bfr[sn][8 + e] = BsC[(kb + 16 + e) * BSN + n];
                }
            }
        } else {
#pragma unroll
            for (int sn = 0; sn < 2; ++sn) {
                const bf16* bp = BsC + (waveN * 32 + sn * 16 + lm) * BST;
                FragU u;
                u.q[0] = *(const uint4*)(bp + kb);
                u.q[1] = *(const uint4*)(bp + kb + 16);
                bfr[sn] = u.v;
            }
        }
#pragma unroll
        for (int sm = 0; sm < 4; ++sm)
#pragma unroll
            for (int sn = 0; sn < 2; ++sn)
                acc[sm][sn] = __builtin_amdgcn_wmma_f32_16x16x32_bf16(
                    false, afr[sm], false, bfr[sn], (short)0, acc[sm][sn], false, false);
        // no bottom barrier: next iteration's wait+barrier orders buffer reuse
    }

    // ---- epilogue: C/D layout VGPR r -> M=r (lanes<16) / M=r+8, N=lane%16
    float* cf = p.Cf ? p.Cf + coff : (float*)0;
    bf16*  cb = p.Cb ? p.Cb + coff : (bf16*)0;
#pragma unroll
    for (int sm = 0; sm < 4; ++sm)
#pragma unroll
        for (int sn = 0; sn < 2; ++sn) {
            int col = n0 + waveN * 32 + sn * 16 + lm;
#pragma unroll
            for (int r = 0; r < 8; ++r) {
                int row = m0 + waveM * 64 + sm * 16 + hi * 8 + r;
                if (row < p.M && col < p.N) {
                    float v = acc[sm][sn][r] * p.alpha;
                    if (biasz) v += biasz[col];
                    if (p.relu && v < 0.0f) v = 0.0f;
                    long long o = (long long)row * p.ldc + col;
                    if (cf) cf[o] = v;
                    if (cb) cb[o] = (bf16)v;
                }
            }
        }
}

// ---------------------------------------------------------------- masked softmax (key axis)
__global__ __launch_bounds__(256) void softmax_kernel(const float* S, bf16* W,
                                                      int Sk, int Kpad, int causal) {
    __shared__ float red[256];
    int q = blockIdx.x;
    long long base = ((long long)blockIdx.y * SQL + q) * 256;
    int t = threadIdx.x;
    float s = -3.0e38f;
    if (t < Sk) {
        s = S[base + t];
        // faithful to reference: where(tril(scores)==0, NEG, scores)
        if (causal && (t > q || s == 0.0f)) s = -1.0e7f;
    }
    red[t] = s; __syncthreads();
    for (int off = 128; off > 0; off >>= 1) { if (t < off) red[t] = fmaxf(red[t], red[t + off]); __syncthreads(); }
    float mx = red[0]; __syncthreads();
    float e = (t < Sk) ? __expf(s - mx) : 0.0f;
    red[t] = e; __syncthreads();
    for (int off = 128; off > 0; off >>= 1) { if (t < off) red[t] += red[t + off]; __syncthreads(); }
    float inv = 1.0f / red[0];
    if (t < Kpad) W[base + t] = (bf16)(e * inv);
}

// ---------------------------------------------------------------- residual + LayerNorm
__global__ __launch_bounds__(256) void add_ln_kernel(const float* X, const float* Y,
                                                     const float* g, const float* b,
                                                     float* outF, bf16* outB) {
    __shared__ float red[256];
    long long base = (long long)blockIdx.x * HD;
    int t = threadIdx.x;
    float v0 = X[base + t] + Y[base + t];
    float v1 = X[base + t + 256] + Y[base + t + 256];
    red[t] = v0 + v1; __syncthreads();
    for (int off = 128; off > 0; off >>= 1) { if (t < off) red[t] += red[t + off]; __syncthreads(); }
    float mean = red[0] * (1.0f / HD); __syncthreads();
    float d0 = v0 - mean, d1 = v1 - mean;
    red[t] = d0 * d0 + d1 * d1; __syncthreads();
    for (int off = 128; off > 0; off >>= 1) { if (t < off) red[t] += red[t + off]; __syncthreads(); }
    float rstd = rsqrtf(red[0] * (1.0f / HD) + 1e-5f);
    float o0 = g[t] * d0 * rstd + b[t];
    float o1 = g[t + 256] * d1 * rstd + b[t + 256];
    outF[base + t] = o0;        outF[base + t + 256] = o1;
    outB[base + t] = (bf16)o0;  outB[base + t + 256] = (bf16)o1;
}

// ---------------------------------------------------------------- embedding + sinusoidal PE
__global__ __launch_bounds__(256) void embed_kernel(const int* ids, const float* emb,
                                                    float* xf, bf16* xb) {
    int row = blockIdx.x;               // 0..TT-1
    int s = row & (SQL - 1);
    long long base  = (long long)row * HD;
    long long ebase = (long long)ids[row] * HD;
    int t = threadIdx.x;
#pragma unroll
    for (int j = 0; j < 2; ++j) {
        int h = t + j * 256;
        float freq = __expf(-9.210340371976184f * (float)((h >> 1) << 1) / (float)HD);
        float ang = (float)s * freq;
        float pe = (h & 1) ? __cosf(ang) : __sinf(ang);
        float v = emb[ebase + h] + pe;
        xf[base + h] = v;
        xb[base + h] = (bf16)v;
    }
}

// ---------------------------------------------------------------- conversions
__global__ void f2bf_kernel(const float* s, bf16* d, long long n) {
    long long i = (long long)blockIdx.x * blockDim.x + threadIdx.x;
    long long st = (long long)gridDim.x * blockDim.x;
    for (; i < n; i += st) d[i] = (bf16)s[i];
}
__global__ void annot_kernel(const float* s, bf16* d) {
    long long n = (long long)BB * 256 * HD;
    long long i = (long long)blockIdx.x * blockDim.x + threadIdx.x;
    long long st = (long long)gridDim.x * blockDim.x;
    for (; i < n; i += st) {
        long long b = i / (256 * HD);
        int r = (int)((i / HD) % 256);
        int h = (int)(i % HD);
        d[i] = (r < SEL) ? (bf16)s[(b * SEL + r) * HD + h] : (bf16)0.0f;
    }
}

// ---------------------------------------------------------------- host-side launch helpers
static void gemm(hipStream_t st, const bf16* A, const bf16* B, const float* bias,
                 float* Cf, bf16* Cb, int M, int N, int K, int lda, int ldb, int ldc,
                 int Z, int zdiv, long long aZd, long long aZm, long long bZd, long long bZm,
                 long long cZd, long long cZm, long long biasZd, long long biasZm,
                 float alpha, int transB, int relu) {
    GemmP p{A, B, bias, Cf, Cb, M, N, K, lda, ldb, ldc, zdiv, transB, relu,
            aZd, aZm, bZd, bZm, cZd, cZm, biasZd, biasZm, alpha};
    dim3 grid((M + BM - 1) / BM, (N + BN - 1) / BN, Z);
    hipLaunchKernelGGL(gemm_bf16_kernel, grid, dim3(256), 0, st, p);
}
static void f2bf(hipStream_t st, const float* s, bf16* d, long long n) {
    hipLaunchKernelGGL(f2bf_kernel, dim3(1024), dim3(256), 0, st, s, d, n);
}

extern "C" void kernel_launch(void* const* d_in, const int* in_sizes, int n_in,
                              void* d_out, int out_size, void* d_ws, size_t ws_size,
                              hipStream_t stream) {
    const int*   inputs = (const int*)  d_in[0];
    const float* annot  = (const float*)d_in[1];
    const float* emb    = (const float*)d_in[2];
    const float* Wq_s = (const float*)d_in[3],  *bq_s = (const float*)d_in[4];
    const float* Wk_s = (const float*)d_in[5],  *bk_s = (const float*)d_in[6];
    const float* Wv_s = (const float*)d_in[7],  *bv_s = (const float*)d_in[8];
    const float* Wq_e = (const float*)d_in[9],  *bq_e = (const float*)d_in[10];
    const float* Wk_e = (const float*)d_in[11], *bk_e = (const float*)d_in[12];
    const float* Wv_e = (const float*)d_in[13], *bv_e = (const float*)d_in[14];
    const float* Wc   = (const float*)d_in[15], *bc   = (const float*)d_in[16];
    const float* Ws   = (const float*)d_in[17], *bs   = (const float*)d_in[18];
    const float* Wm   = (const float*)d_in[19], *bm   = (const float*)d_in[20];
    const float* g1 = (const float*)d_in[21], *be1 = (const float*)d_in[22];
    const float* g2 = (const float*)d_in[23], *be2 = (const float*)d_in[24];
    const float* g3 = (const float*)d_in[25], *be3 = (const float*)d_in[26];
    const float* Wout = (const float*)d_in[27], *bout = (const float*)d_in[28];

    // ---- workspace layout (256B aligned)
    char* wp = (char*)d_ws;
    auto alloc = [&](size_t bytes) -> char* {
        char* p = wp; wp += (bytes + 255) & ~(size_t)255; return p;
    };
    const long long NHHH = (long long)NHD * HD * HD;   // 2,097,152
    float* x_f  = (float*)alloc((size_t)TT * HD * 4);
    bf16*  x_b  = (bf16*) alloc((size_t)TT * HD * 2);
    float* r1_f = (float*)alloc((size_t)TT * HD * 4);
    bf16*  r1_b = (bf16*) alloc((size_t)TT * HD * 2);
    float* r2_f = (float*)alloc((size_t)TT * HD * 4);
    bf16*  r2_b = (bf16*) alloc((size_t)TT * HD * 2);
    bf16*  an_b = (bf16*) alloc((size_t)BB * 256 * HD * 2);
    bf16*  qb   = (bf16*) alloc((size_t)ZB * SLOT * 2);
    bf16*  kbuf = (bf16*) alloc((size_t)ZB * SLOT * 2);
    bf16*  vbuf = (bf16*) alloc((size_t)ZB * SLOT * 2);
    float* sc   = (float*)alloc((size_t)ZB * SQL * 256 * 4);
    bf16*  wsm  = (bf16*) alloc((size_t)ZB * SQL * 256 * 2);
    bf16*  ctx  = (bf16*) alloc((size_t)TT * NHD * HD * 2);
    float* tmp  = (float*)alloc((size_t)TT * HD * 4);
    bf16*  wq_b = (bf16*) alloc((size_t)NHHH * 2);
    bf16*  wk_b = (bf16*) alloc((size_t)NHHH * 2);
    bf16*  wv_b = (bf16*) alloc((size_t)NHHH * 2);
    bf16*  wcs_b= (bf16*) alloc((size_t)NHD * HD * HD * 2);
    bf16*  wm_b = (bf16*) alloc((size_t)HD * HD * 2);
    bf16*  wo_b = (bf16*) alloc((size_t)HD * VOC * 2);

    const float SCALE = 0.044194173824159216f;   // 1/sqrt(512)

    // ---- prologue
    hipLaunchKernelGGL(embed_kernel, dim3(TT), dim3(256), 0, stream, inputs, emb, x_f, x_b);
    hipLaunchKernelGGL(annot_kernel, dim3(1024), dim3(256), 0, stream, annot, an_b);
    f2bf(stream, Wout, wo_b, (long long)HD * VOC);

    for (int i = 0; i < NLAY; ++i) {
        const long long WOFF = (long long)i * NHHH;
        const long long BOFF = (long long)i * NHD * HD;

        // ======== self-attention ========
        f2bf(stream, Wq_s + WOFF, wq_b, NHHH);
        f2bf(stream, Wk_s + WOFF, wk_b, NHHH);
        f2bf(stream, Wv_s + WOFF, wv_b, NHHH);
        // q/k/v = x @ W[n] + b[n], batched over z = b*NH + n
        gemm(stream, x_b, wq_b, bq_s + BOFF, 0, qb,   SQL, HD, HD, HD, HD, HD,
             ZB, NHD, (long long)SQL*HD, 0, 0, (long long)HD*HD,
             (long long)NHD*SLOT, SLOT, 0, HD, 1.0f, 0, 0);
        gemm(stream, x_b, wk_b, bk_s + BOFF, 0, kbuf, SQL, HD, HD, HD, HD, HD,
             ZB, NHD, (long long)SQL*HD, 0, 0, (long long)HD*HD,
             (long long)NHD*SLOT, SLOT, 0, HD, 1.0f, 0, 0);
        gemm(stream, x_b, wv_b, bv_s + BOFF, 0, vbuf, SQL, HD, HD, HD, HD, HD,
             ZB, NHD, (long long)SQL*HD, 0, 0, (long long)HD*HD,
             (long long)NHD*SLOT, SLOT, 0, HD, 1.0f, 0, 0);
        // scores[q,k] = SCALE * q . k   (B transposed)
        gemm(stream, qb, kbuf, 0, sc, 0, SQL, SQL, HD, HD, HD, 256,
             ZB, 1, SLOT, 0, SLOT, 0, (long long)SQL*256, 0, 0, 0, SCALE, 1, 0);
        hipLaunchKernelGGL(softmax_kernel, dim3(SQL, ZB), dim3(256), 0, stream,
                           sc, wsm, SQL, 256, 1);
        // ctx[b, q, n*H+d] = sum_k w[q,k] v[k,d]
        gemm(stream, wsm, vbuf, 0, 0, ctx, SQL, HD, 256, 256, HD, NHD*HD,
             ZB, NHD, (long long)NHD*SQL*256, (long long)SQL*256,
             (long long)NHD*SLOT, SLOT, (long long)SQL*NHD*HD, HD, 0, 0, 1.0f, 0, 0);
        // concat projection + residual + LN
        f2bf(stream, Wc + WOFF, wcs_b, NHHH);
        gemm(stream, ctx, wcs_b, bc + (long long)i*HD, tmp, 0, TT, HD, NHD*HD,
             NHD*HD, HD, HD, 1, 1, 0,0,0,0,0,0,0,0, 1.0f, 0, 0);
        hipLaunchKernelGGL(add_ln_kernel, dim3(TT), dim3(256), 0, stream,
                           x_f, tmp, g1 + (long long)i*HD, be1 + (long long)i*HD, r1_f, r1_b);

        // ======== cross-attention ========
        f2bf(stream, Wq_e + WOFF, wq_b, NHHH);
        f2bf(stream, Wk_e + WOFF, wk_b, NHHH);
        f2bf(stream, Wv_e + WOFF, wv_b, NHHH);
        gemm(stream, r1_b, wq_b, bq_e + BOFF, 0, qb,   SQL, HD, HD, HD, HD, HD,
             ZB, NHD, (long long)SQL*HD, 0, 0, (long long)HD*HD,
             (long long)NHD*SLOT, SLOT, 0, HD, 1.0f, 0, 0);
        gemm(stream, an_b, wk_b, bk_e + BOFF, 0, kbuf, SEP, HD, HD, HD, HD, HD,
             ZB, NHD, (long long)256*HD, 0, 0, (long long)HD*HD,
             (long long)NHD*SLOT, SLOT, 0, HD, 1.0f, 0, 0);
        gemm(stream, an_b, wv_b, bv_e + BOFF, 0, vbuf, SEP, HD, HD, HD, HD, HD,
             ZB, NHD, (long long)256*HD, 0, 0, (long long)HD*HD,
             (long long)NHD*SLOT, SLOT, 0, HD, 1.0f, 0, 0);
        gemm(stream, qb, kbuf, 0, sc, 0, SQL, SEL, HD, HD, HD, 256,
             ZB, 1, SLOT, 0, SLOT, 0, (long long)SQL*256, 0, 0, 0, SCALE, 1, 0);
        hipLaunchKernelGGL(softmax_kernel, dim3(SQL, ZB), dim3(256), 0, stream,
                           sc, wsm, SEL, SEP, 0);
        gemm(stream, wsm, vbuf, 0, 0, ctx, SQL, HD, SEP, 256, HD, NHD*HD,
             ZB, NHD, (long long)NHD*SQL*256, (long long)SQL*256,
             (long long)NHD*SLOT, SLOT, (long long)SQL*NHD*HD, HD, 0, 0, 1.0f, 0, 0);
        f2bf(stream, Ws + WOFF, wcs_b, NHHH);
        gemm(stream, ctx, wcs_b, bs + (long long)i*HD, tmp, 0, TT, HD, NHD*HD,
             NHD*HD, HD, HD, 1, 1, 0,0,0,0,0,0,0,0, 1.0f, 0, 0);
        hipLaunchKernelGGL(add_ln_kernel, dim3(TT), dim3(256), 0, stream,
                           r1_f, tmp, g2 + (long long)i*HD, be2 + (long long)i*HD, r2_f, r2_b);

        // ======== MLP ========
        f2bf(stream, Wm + (long long)i*HD*HD, wm_b, (long long)HD*HD);
        gemm(stream, r2_b, wm_b, bm + (long long)i*HD, tmp, 0, TT, HD, HD,
             HD, HD, HD, 1, 1, 0,0,0,0,0,0,0,0, 1.0f, 0, 1 /*relu*/);
        hipLaunchKernelGGL(add_ln_kernel, dim3(TT), dim3(256), 0, stream,
                           r2_f, tmp, g3 + (long long)i*HD, be3 + (long long)i*HD, x_f, x_b);
    }

    // ---- final vocab projection: (TT,512) @ (512,16000) + bout -> d_out (f32)
    gemm(stream, x_b, wo_b, bout, (float*)d_out, 0, TT, VOC, HD, HD, VOC, VOC,
         1, 1, 0,0,0,0,0,0,0,0, 1.0f, 0, 0);
    (void)in_sizes; (void)n_in; (void)out_size; (void)ws_size;
}
